// QuantLinearBase_41480794145182
// MI455X (gfx1250) — compile-verified, matrix operand order checked by reading, not run on hardware
//
#include <hip/hip_runtime.h>
#include <hip/hip_bf16.h>

typedef __attribute__((ext_vector_type(16))) _Float16 v16h;
typedef __attribute__((ext_vector_type(8)))  _Float16 v8h;
typedef __attribute__((ext_vector_type(4)))  _Float16 v4h;
typedef __attribute__((ext_vector_type(8)))  float    v8f;

#define IN_F    4096
#define OUT_F   11008
#define QZ_COLS (OUT_F / 8)   // 1376
#define M_TOT   8192          // 4 * 2048
#define BM 128
#define BN 128
#define BK 64                 // half a quant group -> group id uniform per K-tile
#define NKT (IN_F / BK)       // 64 K-tiles
#define LDA 72                // BK + 8 halves padding (144 B row stride, 16B aligned)
#define LDB 72
#define A_ELEMS (BM * LDA)    // 9216 halves / buffer
#define B_ELEMS (BN * LDB)

// Per-thread register staging for one K-tile (loaded for tile kt+1 while tile kt computes)
struct Stage {
  float4 xr[8];   // 128x64 fp32 x-tile, 8 float4 per thread
  int    qw[4];   // 8x128 packed qweight words, 4 per thread
  int    zword;   // qzeros word for this thread's column's group
  float  s;       // scale for this thread's column's group
};

// Load a 16x32 f16 WMMA operand fragment from an LDS tile stored row-major with
// stride LD halves. ISA layout (16-bit A 16x32): lanes 0-15 hold K = koff..+7 and
// koff+16..+23; lanes 16-31 hold K offset +8 (caller folds into koff).
__device__ __forceinline__ v16h frag_ld(const _Float16* rowbase, int koff) {
  v8h lo = *reinterpret_cast<const v8h*>(rowbase + koff);
  v8h hi = *reinterpret_cast<const v8h*>(rowbase + koff + 16);
  v16h r;
#pragma unroll
  for (int i = 0; i < 8; ++i) { r[i] = lo[i]; r[i + 8] = hi[i]; }
  return r;
}

__global__ __launch_bounds__(256)
void gptq_gemm_wmma_f16(const float* __restrict__ x,
                        const int*   __restrict__ qweight,
                        const int*   __restrict__ qzeros,
                        const float* __restrict__ scales,
                        const float* __restrict__ bias,
                        float*       __restrict__ out) {
  // double-buffered tiles: 2 * (18432 + 18432) = 73728 B (< 320 KB WGP pool)
  __shared__ alignas(16) _Float16 As[2 * A_ELEMS];  // [m][k]
  __shared__ alignas(16) _Float16 Bs[2 * B_ELEMS];  // [n][k]

  const int tid  = threadIdx.x;
  const int lane = tid & 31;
  const int wave = tid >> 5;

  const int m0 = blockIdx.y * BM;
  const int n0 = blockIdx.x * BN;

  // 8 waves tile the 128x128 block: 4 along M (32 rows), 2 along N (64 cols)
  const int wm = (wave & 3) * 32;
  const int wn = (wave >> 2) * 64;

  // dequant ownership: one column per thread, 4 of the 8 packed rows
  const int dq_c    = tid & 127;
  const int dq_rgrp = (tid >> 7) * 4;
  const int dq_n    = n0 + dq_c;

  v8f acc[2][4];
#pragma unroll
  for (int mt = 0; mt < 2; ++mt)
#pragma unroll
    for (int nt = 0; nt < 4; ++nt)
#pragma unroll
      for (int r = 0; r < 8; ++r) acc[mt][nt][r] = 0.0f;

  // ---------- staging helpers (lambdas keep per-thread indices captured) ----------
  auto stage_load = [&](Stage& st, int kt) {
#pragma unroll
    for (int i = 0; i < 8; ++i) {
      const int idx = i * 256 + tid;
      const int row = idx >> 4;       // 16 float4 per 64-float row
      const int c4  = idx & 15;
      st.xr[i] = reinterpret_cast<const float4*>(
          x + (size_t)(m0 + row) * IN_F + kt * BK)[c4];
    }
    const int g     = kt >> 1;        // GROUPSIZE=128, BK=64
    const int qrow0 = kt * (BK / 8);
#pragma unroll
    for (int j = 0; j < 4; ++j)
      st.qw[j] = qweight[(size_t)(qrow0 + dq_rgrp + j) * OUT_F + dq_n];
    st.zword = qzeros[(size_t)g * QZ_COLS + (dq_n >> 3)];
    st.s     = scales[(size_t)g * OUT_F + dq_n];
  };

  auto stage_store = [&](const Stage& st, int buf) {
    _Float16* Ab = As + buf * A_ELEMS;
    _Float16* Bb = Bs + buf * B_ELEMS;
#pragma unroll
    for (int i = 0; i < 8; ++i) {
      const int idx = i * 256 + tid;
      const int row = idx >> 4;
      const int c4  = idx & 15;
      v4h h;
      h[0] = (_Float16)st.xr[i].x; h[1] = (_Float16)st.xr[i].y;
      h[2] = (_Float16)st.xr[i].z; h[3] = (_Float16)st.xr[i].w;
      *reinterpret_cast<v4h*>(&Ab[row * LDA + c4 * 4]) = h;
    }
    const int   zero = ((st.zword >> ((dq_n & 7) * 4)) & 15) + 1;
    const float s    = st.s;
    const float zb   = -s * (float)zero;   // w = nib*s + zb
#pragma unroll
    for (int j = 0; j < 4; ++j) {
      const int r = dq_rgrp + j;
      const int w = st.qw[j];
      v8h hv;
#pragma unroll
      for (int sft = 0; sft < 8; ++sft) {
        const int nib = (w >> (4 * sft)) & 15;
        hv[sft] = (_Float16)fmaf((float)nib, s, zb);
      }
      *reinterpret_cast<v8h*>(&Bb[dq_c * LDB + r * 8]) = hv;
    }
  };

  // ---------- prologue: stage tile 0 into buffer 0 ----------
  {
    Stage st0;
    stage_load(st0, 0);
    stage_store(st0, 0);
  }
  __syncthreads();

  const int lrow  = lane & 15;
  const int khalf = (lane >> 4) << 3;   // +8 K for upper lane half

  int buf = 0;
  for (int kt = 0; kt < NKT; ++kt) {
    const int nbuf = buf ^ 1;

    // issue global loads for tile kt+1 (no wait yet; latency hidden by WMMAs)
    Stage st;
    if (kt + 1 < NKT) stage_load(st, kt + 1);

    // ---- compute tile kt from LDS buffer `buf`: 2 k-steps x 8 WMMA per wave ----
    const _Float16* Ab = As + buf * A_ELEMS;
    const _Float16* Bb = Bs + buf * B_ELEMS;
#pragma unroll
    for (int ks = 0; ks < 2; ++ks) {
      const int koff = ks * 32 + khalf;
      v16h a0 = frag_ld(&Ab[(wm + lrow) * LDA], koff);
      v16h a1 = frag_ld(&Ab[(wm + 16 + lrow) * LDA], koff);
      v16h b[4];
#pragma unroll
      for (int nt = 0; nt < 4; ++nt)
        b[nt] = frag_ld(&Bb[(wn + nt * 16 + lrow) * LDB], koff);
#pragma unroll
      for (int nt = 0; nt < 4; ++nt) {
        acc[0][nt] = __builtin_amdgcn_wmma_f32_16x16x32_f16(
            false, a0, false, b[nt], (short)0, acc[0][nt], false, false);
        acc[1][nt] = __builtin_amdgcn_wmma_f32_16x16x32_f16(
            false, a1, false, b[nt], (short)0, acc[1][nt], false, false);
      }
    }

    // ---- drain staged registers into the other buffer; single barrier ----
    if (kt + 1 < NKT) stage_store(st, nbuf);
    __syncthreads();
    buf = nbuf;
  }

  // ---- epilogue: D layout -> global, add bias ----
  const int ml = (lane >> 4) * 8;  // lanes 16-31 hold M = r+8
  const int nl = lane & 15;
  float bv[4];
#pragma unroll
  for (int nt = 0; nt < 4; ++nt) bv[nt] = bias[n0 + wn + nt * 16 + nl];

#pragma unroll
  for (int mt = 0; mt < 2; ++mt) {
#pragma unroll
    for (int nt = 0; nt < 4; ++nt) {
      const int n = n0 + wn + nt * 16 + nl;
#pragma unroll
      for (int r = 0; r < 8; ++r) {
        const int m = m0 + wm + mt * 16 + ml + r;
        out[(size_t)m * OUT_F + n] = acc[mt][nt][r] + bv[nt];
      }
    }
  }
}

extern "C" void kernel_launch(void* const* d_in, const int* in_sizes, int n_in,
                              void* d_out, int out_size, void* d_ws, size_t ws_size,
                              hipStream_t stream) {
  (void)in_sizes; (void)n_in; (void)out_size; (void)d_ws; (void)ws_size;
  const float* x       = (const float*)d_in[0];
  const int*   qweight = (const int*)  d_in[1];
  const int*   qzeros  = (const int*)  d_in[2];
  const float* scales  = (const float*)d_in[3];
  // d_in[4] = g_idx: identity mapping k/GROUPSIZE, folded into the kernel
  const float* bias    = (const float*)d_in[5];
  float* out = (float*)d_out;

  dim3 grid(OUT_F / BN, M_TOT / BM);  // (86, 64)
  dim3 block(256);
  gptq_gemm_wmma_f16<<<grid, block, 0, stream>>>(x, qweight, qzeros, scales, bias, out);
}
